// CorrectorEGNN_661424963984
// MI455X (gfx1250) — matrix operand order repaired and checked by reading, not compiled, not run on hardware
//
#include <hip/hip_runtime.h>

// Sizes from the reference
#define NPART 64
#define HD    128
#define NLAY  4
#define NGRAPH 128

// LDS pitches (in elements) chosen to avoid 64-bank conflicts:
// 132 f32 = 528B, 136 f16 = 272B, 264 f16 = 528B  -> row-to-row shift of 4 banks.
#define P32  132
#define P16  136
#define PW1  264

typedef _Float16 v16h __attribute__((ext_vector_type(16)));
typedef float    v8f  __attribute__((ext_vector_type(8)));

__device__ __forceinline__ v16h ld2(const _Float16* p0, const _Float16* p1) {
    union { uint4 u[2]; v16h v; } f;
    f.u[0] = *(const uint4*)p0;
    f.u[1] = *(const uint4*)p1;
    return f.v;
}

// A-fragment (16x32 f16, row-major source): lane half selects K sub-chunks
// {0..7,16..23} (lanes 0-15) / {8..15,24..31} (lanes 16-31). row already points
// at rowstart + kbase.
__device__ __forceinline__ v16h fragA(const _Float16* row, int koff8) {
    return ld2(row + koff8, row + 16 + koff8);
}

// B-fragment (32x16 f16 from W^T stored [n][k]): lanes 0-15 hold K {0..7,16..23},
// lanes 16-31 hold K {8..15,24..31} of the chunk. row points at W^T[n] + kbase.
__device__ __forceinline__ v16h fragB(const _Float16* row, int hi) {
    const _Float16* p = row + hi * 8;
    return ld2(p, p + 16);
}

__device__ __forceinline__ v8f wmma32(v16h a, v16h b, v8f c) {
    return __builtin_amdgcn_wmma_f32_16x16x32_f16(false, a, false, b, (short)0, c, false, false);
}

__device__ __forceinline__ float silu_f(float x) {
    return x * (1.0f / (1.0f + __expf(-x)));
}

// ---- prep: dst[l][n][k] = (f16) src[l][k*Nn + n]  (weight transpose+convert)
__global__ void egnn_wprep(const float* __restrict__ src, _Float16* __restrict__ dst,
                           int Ksrc, int Kdst, int Nn, int total) {
    int i = blockIdx.x * 256 + threadIdx.x;
    if (i >= total) return;
    int per = Nn * Kdst;
    int l = i / per;
    int r = i - l * per;
    int n = r / Kdst;
    int k = r - n * Kdst;
    dst[i] = (_Float16)src[(l * Ksrc + k) * Nn + n];
}

// ---- main fused kernel: one workgroup (256 threads = 8 waves) per graph
__global__ __launch_bounds__(256) void egnn_main(
    const float* __restrict__ x,
    const float* __restrict__ node_embed,
    const float* __restrict__ output_scale,
    const float* __restrict__ edge_w1, const float* __restrict__ edge_b1,
    const float* __restrict__ edge_b2,
    const float* __restrict__ coord_b1, const float* __restrict__ coord_w2,
    const float* __restrict__ node_b1, const float* __restrict__ node_b2,
    const _Float16* __restrict__ w1t,  // [L][128][256]
    const _Float16* __restrict__ w2t,  // [L][128][128]
    const _Float16* __restrict__ cw1t, // [L][128][128]
    const _Float16* __restrict__ nw1t, // [L][128][256]
    const _Float16* __restrict__ nw2t, // [L][128][128]
    float* __restrict__ out)
{
    __shared__ float    sh_h[NPART * P32];
    __shared__ _Float16 sh_h16[NPART * P16];
    __shared__ float    sh_msg[NPART * P32];
    __shared__ _Float16 sh_msg16[NPART * P16];
    __shared__ float    sh_pos[NPART][4];
    __shared__ float    sh_dpos[NPART][4];
    __shared__ _Float16 sh_w1t[HD * PW1];
    __shared__ _Float16 sh_w2t[HD * P16];
    __shared__ _Float16 sh_cw1t[HD * P16];
    __shared__ float    sh_b1[HD], sh_b2[HD], sh_cb1[HD], sh_cw2[HD], sh_w1last[HD];
    __shared__ _Float16 sh_stage[8][16 * P16];   // per-wave A staging
    __shared__ float    sh_rel[8][16][4];        // per-wave rel.xyz, d2
    __shared__ float    sh_mean[4];

    const int tid  = threadIdx.x;
    const int wv   = tid >> 5;
    const int lane = tid & 31;
    const int hi   = lane >> 4;    // which lane half
    const int lo   = lane & 15;
    const int koff8 = hi * 8;
    const int b = blockIdx.x;

    // ---- init: pos from x, h from node_embed
    float p0x = 0.f, p0y = 0.f, p0z = 0.f;
    if (tid < NPART) {
        int base = (b * NPART + tid) * 3;
        p0x = x[base + 0]; p0y = x[base + 1]; p0z = x[base + 2];
        sh_pos[tid][0] = p0x; sh_pos[tid][1] = p0y; sh_pos[tid][2] = p0z; sh_pos[tid][3] = 0.f;
    }
    for (int i = tid; i < NPART * HD; i += 256) {
        int r = i >> 7, c = i & 127;
        float v = node_embed[c];
        sh_h[r * P32 + c] = v;
        sh_h16[r * P16 + c] = (_Float16)v;
    }
    __syncthreads();

    for (int l = 0; l < NLAY; ++l) {
        // ---- load this layer's edge weights into LDS (f16, transposed)
        const _Float16* gw1 = w1t  + l * HD * 256;
        const _Float16* gw2 = w2t  + l * HD * HD;
        const _Float16* gc1 = cw1t + l * HD * HD;
        for (int i = tid; i < HD * 32; i += 256) {           // w1t: 128 rows x 32 uint4
            int n = i >> 5, c8 = i & 31;
            ((uint4*)(sh_w1t + n * PW1))[c8] = ((const uint4*)(gw1 + n * 256))[c8];
        }
        for (int i = tid; i < HD * 16; i += 256) {           // w2t / cw1t: 128 x 16 uint4
            int n = i >> 4, c8 = i & 15;
            ((uint4*)(sh_w2t  + n * P16))[c8] = ((const uint4*)(gw2 + n * HD))[c8];
            ((uint4*)(sh_cw1t + n * P16))[c8] = ((const uint4*)(gc1 + n * HD))[c8];
        }
        if (tid < HD) {
            sh_b1[tid]     = edge_b1[l * HD + tid];
            sh_b2[tid]     = edge_b2[l * HD + tid];
            sh_cb1[tid]    = coord_b1[l * HD + tid];
            sh_cw2[tid]    = coord_w2[l * HD + tid];
            sh_w1last[tid] = edge_w1[(l * 257 + 256) * HD + tid];   // d^2 row of W1
        }
        for (int i = tid; i < NPART * P32; i += 256) sh_msg[i] = 0.f;
        if (tid < NPART) { sh_dpos[tid][0] = 0.f; sh_dpos[tid][1] = 0.f; sh_dpos[tid][2] = 0.f; sh_dpos[tid][3] = 0.f; }
        __syncthreads();

        // ---- edge phase: 256 tiles of (dst, 16 srcs); 32 tiles per wave
        _Float16* stg = sh_stage[wv];
        for (int t = wv; t < 256; t += 8) {
            const int dst = t >> 2;
            const int s0  = (t & 3) << 4;
            const int selfRow = (dst >= s0 && dst < s0 + 16) ? (dst - s0) : -1;

            if (lane < 16) {
                int src = s0 + lane;
                float rx = sh_pos[src][0] - sh_pos[dst][0];
                float ry = sh_pos[src][1] - sh_pos[dst][1];
                float rz = sh_pos[src][2] - sh_pos[dst][2];
                sh_rel[wv][lane][0] = rx; sh_rel[wv][lane][1] = ry;
                sh_rel[wv][lane][2] = rz; sh_rel[wv][lane][3] = rx * rx + ry * ry + rz * rz;
            }
            float d2r[8];
#pragma unroll
            for (int j = 0; j < 8; ++j) d2r[j] = sh_rel[wv][j + hi * 8][3];

            // -- stage 1: t1 = silu(e_in @ W1 + b1), K = 2H (+ d2 rank-1)
            //    C starts as inline 0; bias folded into epilogue.
            v8f acc[8];
#pragma unroll
            for (int p = 0; p < 8; ++p) acc[p] = (v8f){};
#pragma unroll
            for (int c = 0; c < 8; ++c) {
                const _Float16* ar = (c < 4)
                    ? (sh_h16 + (s0 + lo) * P16 + c * 32)          // h[src] channels
                    : (sh_h16 + dst * P16 + (c - 4) * 32);         // h[dst] channels (broadcast)
                v16h af = fragA(ar, koff8);
#pragma unroll
                for (int p = 0; p < 8; ++p) {
                    v16h bf = fragB(sh_w1t + (p * 16 + lo) * PW1 + c * 32, hi);
                    acc[p] = wmma32(af, bf, acc[p]);
                }
            }
#pragma unroll
            for (int p = 0; p < 8; ++p) {
                float wl = sh_w1last[p * 16 + lo];
                float bv = sh_b1[p * 16 + lo];
#pragma unroll
                for (int j = 0; j < 8; ++j) {
                    float v = silu_f(acc[p][j] + bv + d2r[j] * wl);
                    stg[(j + hi * 8) * P16 + p * 16 + lo] = (_Float16)v;
                }
            }

            // -- stage 2: m = silu(t1 @ W2 + b2)   (reads all of stg, then rewrites)
#pragma unroll
            for (int p = 0; p < 8; ++p) acc[p] = (v8f){};
#pragma unroll
            for (int c = 0; c < 4; ++c) {
                v16h af = fragA(stg + lo * P16 + c * 32, koff8);
#pragma unroll
                for (int p = 0; p < 8; ++p) {
                    v16h bf = fragB(sh_w2t + (p * 16 + lo) * P16 + c * 32, hi);
                    acc[p] = wmma32(af, bf, acc[p]);
                }
            }
#pragma unroll
            for (int p = 0; p < 8; ++p) {
                float bv = sh_b2[p * 16 + lo];
                float vv[8];
                float part = 0.f;
#pragma unroll
                for (int j = 0; j < 8; ++j) {
                    float v = silu_f(acc[p][j] + bv);
                    if (j + hi * 8 == selfRow) v = 0.f;            // drop self edge
                    vv[j] = v;
                    part += v;
                }
                part += __shfl_xor(part, 16, 32);                  // sum both lane halves
                if (lane < 16) atomicAdd(&sh_msg[dst * P32 + p * 16 + lo], part);
#pragma unroll
                for (int j = 0; j < 8; ++j)
                    stg[(j + hi * 8) * P16 + p * 16 + lo] = (_Float16)vv[j];
            }

            // -- stage 3+4: cw = silu(m @ cw1 + cb1) @ cw2  (c-outer: A loaded once per chunk)
#pragma unroll
            for (int p = 0; p < 8; ++p) acc[p] = (v8f){};
#pragma unroll
            for (int c = 0; c < 4; ++c) {
                v16h af = fragA(stg + lo * P16 + c * 32, koff8);
#pragma unroll
                for (int p = 0; p < 8; ++p) {
                    v16h bf = fragB(sh_cw1t + (p * 16 + lo) * P16 + c * 32, hi);
                    acc[p] = wmma32(af, bf, acc[p]);
                }
            }
            float cwp[8];
#pragma unroll
            for (int j = 0; j < 8; ++j) cwp[j] = 0.f;
#pragma unroll
            for (int p = 0; p < 8; ++p) {
                float cb = sh_cb1[p * 16 + lo];
                float c2 = sh_cw2[p * 16 + lo];
#pragma unroll
                for (int j = 0; j < 8; ++j) cwp[j] += silu_f(acc[p][j] + cb) * c2;
            }
#pragma unroll
            for (int m = 1; m < 16; m <<= 1)
#pragma unroll
                for (int j = 0; j < 8; ++j) cwp[j] += __shfl_xor(cwp[j], m, 32);
            if (lo == 0) {                                         // lanes 0 and 16
                float sx = 0.f, sy = 0.f, sz = 0.f;
#pragma unroll
                for (int j = 0; j < 8; ++j) {
                    int r = j + hi * 8;
                    float cw = cwp[j];
                    sx += sh_rel[wv][r][0] * cw;
                    sy += sh_rel[wv][r][1] * cw;
                    sz += sh_rel[wv][r][2] * cw;
                }
                atomicAdd(&sh_dpos[dst][0], sx);
                atomicAdd(&sh_dpos[dst][1], sy);
                atomicAdd(&sh_dpos[dst][2], sz);
            }
        }
        __syncthreads();

        // ---- msg -> f16, pos += dpos
        for (int i = tid; i < NPART * HD; i += 256) {
            int r = i >> 7, c = i & 127;
            sh_msg16[r * P16 + c] = (_Float16)sh_msg[r * P32 + c];
        }
        if (tid < NPART) {
            sh_pos[tid][0] += sh_dpos[tid][0];
            sh_pos[tid][1] += sh_dpos[tid][1];
            sh_pos[tid][2] += sh_dpos[tid][2];
        }
        __syncthreads();

        // ---- node phase: h += silu([h|msg] @ nW1 + nb1) @ nW2 + nb2 (waves 0..3)
        if (wv < 4) {
            const int rowbase = wv * 16;
            const _Float16* gn1 = nw1t + l * HD * 256;
            const _Float16* gn2 = nw2t + l * HD * HD;
            v8f acc[8];
#pragma unroll
            for (int p = 0; p < 8; ++p) acc[p] = (v8f){};
#pragma unroll
            for (int c = 0; c < 8; ++c) {
                const _Float16* ar = (c < 4)
                    ? (sh_h16   + (rowbase + lo) * P16 + c * 32)
                    : (sh_msg16 + (rowbase + lo) * P16 + (c - 4) * 32);
                v16h af = fragA(ar, koff8);
#pragma unroll
                for (int p = 0; p < 8; ++p) {
                    v16h bf = fragB(gn1 + (p * 16 + lo) * 256 + c * 32, hi);
                    acc[p] = wmma32(af, bf, acc[p]);
                }
            }
#pragma unroll
            for (int p = 0; p < 8; ++p) {
                float bv = node_b1[l * HD + p * 16 + lo];
#pragma unroll
                for (int j = 0; j < 8; ++j)
                    stg[(j + hi * 8) * P16 + p * 16 + lo] = (_Float16)silu_f(acc[p][j] + bv);
            }
#pragma unroll
            for (int p = 0; p < 8; ++p) acc[p] = (v8f){};
#pragma unroll
            for (int c = 0; c < 4; ++c) {
                v16h af = fragA(stg + lo * P16 + c * 32, koff8);
#pragma unroll
                for (int p = 0; p < 8; ++p) {
                    v16h bf = fragB(gn2 + (p * 16 + lo) * HD + c * 32, hi);
                    acc[p] = wmma32(af, bf, acc[p]);
                }
            }
#pragma unroll
            for (int p = 0; p < 8; ++p) {
                float bv = node_b2[l * HD + p * 16 + lo];
#pragma unroll
                for (int j = 0; j < 8; ++j) {
                    int r = rowbase + j + hi * 8;
                    int cc = p * 16 + lo;
                    float nh = sh_h[r * P32 + cc] + acc[p][j] + bv;
                    sh_h[r * P32 + cc] = nh;
                    sh_h16[r * P16 + cc] = (_Float16)nh;
                }
            }
        }
        __syncthreads();
    }

    // ---- output: dx = pos - pos0, subtract per-graph mean, scale
    if (tid < NPART) {
        sh_dpos[tid][0] = sh_pos[tid][0] - p0x;
        sh_dpos[tid][1] = sh_pos[tid][1] - p0y;
        sh_dpos[tid][2] = sh_pos[tid][2] - p0z;
    }
    __syncthreads();
    if (tid < 3) {
        float s = 0.f;
        for (int n = 0; n < NPART; ++n) s += sh_dpos[n][tid];
        sh_mean[tid] = s * (1.0f / (float)NPART);
    }
    __syncthreads();
    if (tid < NPART) {
        float sc = output_scale[0];
        int base = (b * NPART + tid) * 3;
        out[base + 0] = (sh_dpos[tid][0] - sh_mean[0]) * sc;
        out[base + 1] = (sh_dpos[tid][1] - sh_mean[1]) * sc;
        out[base + 2] = (sh_dpos[tid][2] - sh_mean[2]) * sc;
    }
}

extern "C" void kernel_launch(void* const* d_in, const int* in_sizes, int n_in,
                              void* d_out, int out_size, void* d_ws, size_t ws_size,
                              hipStream_t stream) {
    const float* x            = (const float*)d_in[0];
    const float* node_embed   = (const float*)d_in[1];
    const float* output_scale = (const float*)d_in[2];
    const float* edge_w1      = (const float*)d_in[3];
    const float* edge_b1      = (const float*)d_in[4];
    const float* edge_w2      = (const float*)d_in[5];
    const float* edge_b2      = (const float*)d_in[6];
    const float* coord_w1     = (const float*)d_in[7];
    const float* coord_b1     = (const float*)d_in[8];
    const float* coord_w2     = (const float*)d_in[9];
    const float* node_w1      = (const float*)d_in[10];
    const float* node_b1      = (const float*)d_in[11];
    const float* node_w2      = (const float*)d_in[12];
    const float* node_b2      = (const float*)d_in[13];
    // d_in[14] = edge_index: structure is known (fully connected), not needed.
    float* out = (float*)d_out;

    // workspace layout (f16 elements): pre-transposed weights, ~918 KB total
    _Float16* ws   = (_Float16*)d_ws;
    _Float16* w1t  = ws;                    // 4*128*256
    _Float16* w2t  = ws + 131072;           // 4*128*128
    _Float16* cw1t = ws + 196608;           // 4*128*128
    _Float16* nw1t = ws + 262144;           // 4*128*256
    _Float16* nw2t = ws + 393216;           // 4*128*128

    // weight prep (tiny)
    egnn_wprep<<<(131072 + 255) / 256, 256, 0, stream>>>(edge_w1,  w1t,  257, 256, 128, 131072);
    egnn_wprep<<<(65536  + 255) / 256, 256, 0, stream>>>(edge_w2,  w2t,  128, 128, 128, 65536);
    egnn_wprep<<<(65536  + 255) / 256, 256, 0, stream>>>(coord_w1, cw1t, 128, 128, 128, 65536);
    egnn_wprep<<<(131072 + 255) / 256, 256, 0, stream>>>(node_w1,  nw1t, 256, 256, 128, 131072);
    egnn_wprep<<<(65536  + 255) / 256, 256, 0, stream>>>(node_w2,  nw2t, 128, 128, 128, 65536);

    // one workgroup per graph
    egnn_main<<<NGRAPH, 256, 0, stream>>>(
        x, node_embed, output_scale,
        edge_w1, edge_b1, edge_b2, coord_b1, coord_w2, node_b1, node_b2,
        w1t, w2t, cw1t, nw1t, nw2t, out);
}
